// ViTVelocity_87960930222836
// MI455X (gfx1250) — matrix operand order repaired and checked by reading, not compiled, not run on hardware
//
#include <hip/hip_runtime.h>
#include <hip/hip_bf16.h>
#include <cstdint>

typedef __attribute__((ext_vector_type(16))) __bf16 v16bf;
typedef __attribute__((ext_vector_type(8)))  float  v8f;

union ABf { uint32_t u[8]; v16bf v; };

__device__ __forceinline__ unsigned short f2bf(float x) {
    uint32_t u = __float_as_uint(x);
    uint32_t r = (u + 0x7FFFu + ((u >> 16) & 1u)) >> 16;
    return (unsigned short)r;
}

__device__ __forceinline__ uint32_t f2bf2(float lo, float hi) {
    return (uint32_t)f2bf(lo) | ((uint32_t)f2bf(hi) << 16);
}

__device__ __forceinline__ v8f wmma_bf16(const ABf& a, const ABf& b, v8f c) {
    return __builtin_amdgcn_wmma_f32_16x16x32_bf16(false, a.v, false, b.v,
                                                   (short)0, c, false, false);
}

// async global -> LDS 16B copy (CDNA5 GLOBAL_LOAD_ASYNC_TO_LDS_B128, ASYNCcnt)
__device__ __forceinline__ void async_cp16(void* lds, const void* g) {
    uint32_t lo = (uint32_t)(uintptr_t)lds;          // low 32b of flat = LDS offset
    uint64_t ga = (uint64_t)(uintptr_t)g;
    asm volatile("global_load_async_to_lds_b128 %0, %1, off"
                 :: "v"(lo), "v"(ga) : "memory");
}
__device__ __forceinline__ void wait_async() {
    asm volatile("s_wait_asynccnt 0" ::: "memory");
}

// ---------------------------------------------------------------- constants
#define TOK   3072     // 4 packs * 768 tokens
#define NHEAD 12

// ---------------------------------------------------------------- token meta
__global__ void tok_meta_kernel(const int* __restrict__ patch_hw,
                                const int* __restrict__ packs,
                                int* __restrict__ meta) {
    int g = blockIdx.x * blockDim.x + threadIdx.x;
    if (g >= TOK) return;
    int b = g / 768, j = g % 768;
    int off = 0, idx = 0, ph = 16, pw = 16, jj = 0;
    for (int ii = 0; ii < 4; ++ii) {
        int im = packs[b * 4 + ii];
        int h = patch_hw[im * 2], w = patch_hw[im * 2 + 1];
        int L = h * w;
        if (j < off + L) { idx = im; ph = h; pw = w; jj = j - off; break; }
        off += L;
    }
    meta[g * 4 + 0] = idx; meta[g * 4 + 1] = ph;
    meta[g * 4 + 2] = pw;  meta[g * 4 + 3] = jj;
}

__global__ void zero_kernel(float* __restrict__ p, int n) {
    for (int i = blockIdx.x * blockDim.x + threadIdx.x; i < n; i += gridDim.x * blockDim.x)
        p[i] = 0.f;
}

__global__ __launch_bounds__(256) void patch_gather_kernel(
        const float* __restrict__ images, const int* __restrict__ meta,
        unsigned short* __restrict__ PATCH) {
    int g = blockIdx.x;
    int idx = meta[g * 4], pw = meta[g * 4 + 2], jj = meta[g * 4 + 3];
    int py = jj / pw, px = jj % pw;
    #pragma unroll
    for (int i = 0; i < 3; ++i) {
        int k = threadIdx.x + i * 256;
        int c = k >> 8, rr = (k >> 4) & 15, cc = k & 15;
        float v = images[((size_t)(idx * 3 + c) * 256 + py * 16 + rr) * 256 + px * 16 + cc];
        PATCH[(size_t)g * 768 + k] = f2bf(v);
    }
}

__global__ __launch_bounds__(256) void embed_add_kernel(
        float* __restrict__ X, const int* __restrict__ meta,
        const float* __restrict__ t) {
    int g = blockIdx.x;
    int idx = meta[4 * g], ph = meta[4 * g + 1], pw = meta[4 * g + 2], jj = meta[4 * g + 3];
    int py = jj / pw, px = jj % pw;
    float tv = t[idx];
    const float PI = 3.14159265358979323846f;
    const float LOG1E4 = 9.21034037197618f;
    #pragma unroll
    for (int i = 0; i < 3; ++i) {
        int d = threadIdx.x + i * 256;
        float pos;
        if (d < 384) {
            float fy = (py + 0.5f) / (float)ph;
            int q = (d < 192) ? d : d - 192;
            float ang = fy * exp2f((float)q) * PI;
            pos = (d < 192) ? sinf(ang) : cosf(ang);
        } else {
            int dd = d - 384;
            float fx = (px + 0.5f) / (float)pw;
            int q = (dd < 192) ? dd : dd - 192;
            float ang = fx * exp2f((float)q) * PI;
            pos = (dd < 192) ? sinf(ang) : cosf(ang);
        }
        float te;
        if (d < 384) te = sinf(tv * expf(-LOG1E4 * (float)d / 384.f));
        else         te = cosf(tv * expf(-LOG1E4 * (float)(d - 384) / 384.f));
        X[(size_t)g * 768 + d] += pos + te;
    }
}

__global__ __launch_bounds__(256) void ln_kernel(
        const float* __restrict__ X, const float* __restrict__ w,
        const float* __restrict__ b, unsigned short* __restrict__ Y) {
    __shared__ float red[256];
    int g = blockIdx.x, tid = threadIdx.x;
    const float* x = X + (size_t)g * 768;
    float xs[3];
    #pragma unroll
    for (int i = 0; i < 3; ++i) xs[i] = x[tid + i * 256];
    float s = xs[0] + xs[1] + xs[2];
    red[tid] = s; __syncthreads();
    for (int st = 128; st > 0; st >>= 1) { if (tid < st) red[tid] += red[tid + st]; __syncthreads(); }
    float mu = red[0] * (1.f / 768.f); __syncthreads();
    float v = 0.f;
    #pragma unroll
    for (int i = 0; i < 3; ++i) { float d = xs[i] - mu; v += d * d; }
    red[tid] = v; __syncthreads();
    for (int st = 128; st > 0; st >>= 1) { if (tid < st) red[tid] += red[tid + st]; __syncthreads(); }
    float inv = rsqrtf(red[0] * (1.f / 768.f) + 1e-5f);
    #pragma unroll
    for (int i = 0; i < 3; ++i) {
        int idx = tid + i * 256;
        Y[(size_t)g * 768 + idx] = f2bf((xs[i] - mu) * inv * w[idx] + b[idx]);
    }
}

// ---------------------------------------------------------------- GEMM
// C[M,N] = A(bf16)[M,K] @ B(fp32)[K,N] + bias.  Double-buffered K pipeline:
// A tile via GLOBAL_LOAD_ASYNC_TO_LDS_B128, B tile fp32->bf16 via VGPRs.
// MODE 0: fp32 store  1: bf16 store  2: gelu->bf16  3: fp32 += (residual)
template<int MODE>
__global__ __launch_bounds__(256) void gemm_bf16(
        const unsigned short* __restrict__ A, const float* __restrict__ B,
        const float* __restrict__ bias, float* __restrict__ outF,
        unsigned short* __restrict__ outB, int M, int N, int K) {
    __shared__ __align__(16) unsigned short At[2][128 * 40];
    __shared__ __align__(16) unsigned short Bt[2][128 * 40];
    const int tid = threadIdx.x, wid = tid >> 5, lane = tid & 31;
    const int bm = blockIdx.y * 128, bn = blockIdx.x * 128;
    const int wm = (wid & 1) * 64, wn = (wid >> 1) * 32;
    const int l15 = lane & 15;
    const int kb8 = (lane >> 4) << 3, kb16 = (lane >> 4) << 4;
    const int arow = tid >> 2, aq = (tid & 3) << 3;            // A-copy coords
    const int bk = tid >> 5, bn4 = (tid & 31) << 2;            // B-copy coords

    v8f acc[4][2];
    #pragma unroll
    for (int mt = 0; mt < 4; ++mt)
        #pragma unroll
        for (int nt = 0; nt < 2; ++nt)
            #pragma unroll
            for (int r = 0; r < 8; ++r) acc[mt][nt][r] = 0.f;

    float4 bv[4];
    // ---- prologue: tile 0
    #pragma unroll
    for (int h = 0; h < 2; ++h)
        async_cp16(&At[0][(arow + h * 64) * 40 + aq],
                   A + (size_t)(bm + arow + h * 64) * K + aq);
    #pragma unroll
    for (int i = 0; i < 4; ++i)
        bv[i] = *(const float4*)(B + (size_t)(bk + i * 8) * N + bn + bn4);
    #pragma unroll
    for (int i = 0; i < 4; ++i) {
        const int k = bk + i * 8;
        Bt[0][(bn4 + 0) * 40 + k] = f2bf(bv[i].x);
        Bt[0][(bn4 + 1) * 40 + k] = f2bf(bv[i].y);
        Bt[0][(bn4 + 2) * 40 + k] = f2bf(bv[i].z);
        Bt[0][(bn4 + 3) * 40 + k] = f2bf(bv[i].w);
    }
    wait_async();
    __syncthreads();

    const int nk = K >> 5;
    for (int it = 0; it < nk; ++it) {
        const int buf = it & 1;
        const bool more = (it + 1) < nk;
        if (more) {
            const int kb = (it + 1) << 5;
            #pragma unroll
            for (int h = 0; h < 2; ++h)
                async_cp16(&At[buf ^ 1][(arow + h * 64) * 40 + aq],
                           A + (size_t)(bm + arow + h * 64) * K + kb + aq);
            #pragma unroll
            for (int i = 0; i < 4; ++i)
                bv[i] = *(const float4*)(B + (size_t)(kb + bk + i * 8) * N + bn + bn4);
        }

        ABf aF[4], bF[2];
        #pragma unroll
        for (int mt = 0; mt < 4; ++mt) {
            const int r = wm + mt * 16 + l15;
            #pragma unroll
            for (int p = 0; p < 8; ++p) {
                const int k = ((p >> 2) << 4) + kb8 + ((p & 3) << 1);
                aF[mt].u[p] = *(const uint32_t*)&At[buf][r * 40 + k];
            }
        }
        #pragma unroll
        for (int nt = 0; nt < 2; ++nt) {
            const int r = wn + nt * 16 + l15;
            #pragma unroll
            for (int p = 0; p < 8; ++p)
                bF[nt].u[p] = *(const uint32_t*)&Bt[buf][r * 40 + ((p >> 2) << 4) + kb8 + ((p & 3) << 1)];
        }
        #pragma unroll
        for (int mt = 0; mt < 4; ++mt)
            #pragma unroll
            for (int nt = 0; nt < 2; ++nt)
                acc[mt][nt] = wmma_bf16(aF[mt], bF[nt], acc[mt][nt]);

        if (more) {
            #pragma unroll
            for (int i = 0; i < 4; ++i) {
                const int k = bk + i * 8;
                Bt[buf ^ 1][(bn4 + 0) * 40 + k] = f2bf(bv[i].x);
                Bt[buf ^ 1][(bn4 + 1) * 40 + k] = f2bf(bv[i].y);
                Bt[buf ^ 1][(bn4 + 2) * 40 + k] = f2bf(bv[i].z);
                Bt[buf ^ 1][(bn4 + 3) * 40 + k] = f2bf(bv[i].w);
            }
        }
        wait_async();
        __syncthreads();
    }

    const int rl = (lane >> 4) << 3, cl = lane & 15;
    #pragma unroll
    for (int nt = 0; nt < 2; ++nt) {
        const int cg = bn + wn + nt * 16 + cl;
        const float bvv = bias[cg];
        #pragma unroll
        for (int mt = 0; mt < 4; ++mt) {
            const int rg0 = bm + wm + mt * 16 + rl;
            #pragma unroll
            for (int r = 0; r < 8; ++r) {
                const float val = acc[mt][nt][r] + bvv;
                const size_t o = (size_t)(rg0 + r) * N + cg;
                if (MODE == 0)      outF[o] = val;
                else if (MODE == 1) outB[o] = f2bf(val);
                else if (MODE == 2) outB[o] = f2bf(0.5f * val * (1.f + erff(val * 0.70710678118654752f)));
                else                outF[o] += val;
            }
        }
    }
}

// ---------------------------------------------------------------- attention
__global__ __launch_bounds__(256) void attn_kernel(
        const unsigned short* __restrict__ QKV, unsigned short* __restrict__ AO) {
    __shared__ float Sf[128 * 64];
    __shared__ unsigned short Pb[128 * 64];
    __shared__ unsigned short vT[64 * 64];
    __shared__ float rowM[128], rowL[128], rowF[128], pscr[256];

    const int qblk = blockIdx.x, head = blockIdx.y, pk = blockIdx.z;
    const int tokbase = pk * 768;
    const int tid = threadIdx.x, wid = tid >> 5, lane = tid & 31;
    const int l15 = lane & 15, half = lane >> 4;
    const int kb8 = half << 3;

    if (tid < 128) { rowM[tid] = -3.0e38f; rowL[tid] = 0.f; rowF[tid] = 0.f; }

    ABf qF[2];
    {
        const int qrow = tokbase + qblk * 128 + wid * 16 + l15;
        const unsigned short* qp = QKV + (size_t)qrow * 2304 + head * 64;
        #pragma unroll
        for (int ks = 0; ks < 2; ++ks)
            #pragma unroll
            for (int p = 0; p < 8; ++p) {
                const int k = ks * 32 + ((p >> 2) << 4) + kb8 + ((p & 3) << 1);
                qF[ks].u[p] = *(const uint32_t*)(qp + k);
            }
    }
    v8f accO[4];
    #pragma unroll
    for (int nt = 0; nt < 4; ++nt)
        #pragma unroll
        for (int r = 0; r < 8; ++r) accO[nt][r] = 0.f;
    __syncthreads();

    const int vkt = tid >> 5, vh2 = tid & 31;   // v-transpose coords
    for (int kb = 0; kb < 12; ++kb) {
        const int kt0 = kb * 64;
        #pragma unroll
        for (int i = 0; i < 8; ++i) {
            const int kt = vkt + i * 8;
            const uint32_t val = *(const uint32_t*)(QKV
                + (size_t)(tokbase + kt0 + kt) * 2304 + 1536 + head * 64 + vh2 * 2);
            vT[(vh2 * 2 + 0) * 64 + kt] = (unsigned short)(val & 0xFFFFu);
            vT[(vh2 * 2 + 1) * 64 + kt] = (unsigned short)(val >> 16);
        }

        v8f accS[4];
        #pragma unroll
        for (int nt = 0; nt < 4; ++nt)
            #pragma unroll
            for (int r = 0; r < 8; ++r) accS[nt][r] = 0.f;
        #pragma unroll
        for (int ks = 0; ks < 2; ++ks) {
            ABf bF[4];
            #pragma unroll
            for (int nt = 0; nt < 4; ++nt) {
                const int ktok = tokbase + kt0 + nt * 16 + l15;
                const unsigned short* kp = QKV + (size_t)ktok * 2304 + 768 + head * 64
                                           + kb8 + ks * 32;
                #pragma unroll
                for (int p = 0; p < 8; ++p)
                    bF[nt].u[p] = *(const uint32_t*)(kp + ((p >> 2) << 4) + ((p & 3) << 1));
            }
            #pragma unroll
            for (int nt = 0; nt < 4; ++nt) accS[nt] = wmma_bf16(qF[ks], bF[nt], accS[nt]);
        }
        {
            const int rl = wid * 16 + (half << 3), cl = l15;
            #pragma unroll
            for (int nt = 0; nt < 4; ++nt)
                #pragma unroll
                for (int r = 0; r < 8; ++r)
                    Sf[(rl + r) * 64 + nt * 16 + cl] = accS[nt][r] * 0.125f;
        }
        __syncthreads();

        // online softmax: 2 lanes per row, 32 cols each, float4 vectorized
        const int row = wid * 16 + l15, c0 = half * 32;
        float mx = -3.0e38f;
        #pragma unroll
        for (int cc = 0; cc < 8; ++cc) {
            const float4 s4 = *(const float4*)&Sf[row * 64 + c0 + cc * 4];
            mx = fmaxf(mx, fmaxf(fmaxf(s4.x, s4.y), fmaxf(s4.z, s4.w)));
        }
        pscr[tid] = mx; __syncthreads();
        const float m_blk = fmaxf(pscr[wid * 32 + l15], pscr[wid * 32 + 16 + l15]);
        const float m_old = rowM[row];
        const float m_new = fmaxf(m_old, m_blk);
        float ssum = 0.f;
        #pragma unroll
        for (int cc = 0; cc < 8; ++cc) {
            const float4 s4 = *(const float4*)&Sf[row * 64 + c0 + cc * 4];
            const float e0 = expf(s4.x - m_new), e1 = expf(s4.y - m_new);
            const float e2 = expf(s4.z - m_new), e3 = expf(s4.w - m_new);
            *(uint32_t*)&Pb[row * 64 + c0 + cc * 4]     = f2bf2(e0, e1);
            *(uint32_t*)&Pb[row * 64 + c0 + cc * 4 + 2] = f2bf2(e2, e3);
            ssum += (e0 + e1) + (e2 + e3);
        }
        pscr[tid] = ssum; __syncthreads();
        if (half == 0) {
            const float l_blk = pscr[wid * 32 + l15] + pscr[wid * 32 + 16 + l15];
            const float fac = expf(m_old - m_new);
            rowF[row] = fac;
            rowL[row] = rowL[row] * fac + l_blk;
            rowM[row] = m_new;
        }
        __syncthreads();

        #pragma unroll
        for (int nt = 0; nt < 4; ++nt)
            #pragma unroll
            for (int r = 0; r < 8; ++r)
                accO[nt][r] *= rowF[wid * 16 + (half << 3) + r];
        #pragma unroll
        for (int ks = 0; ks < 2; ++ks) {
            ABf aP;
            const int prow = wid * 16 + l15;
            #pragma unroll
            for (int p = 0; p < 8; ++p) {
                const int k = ks * 32 + ((p >> 2) << 4) + kb8 + ((p & 3) << 1);
                aP.u[p] = *(const uint32_t*)&Pb[prow * 64 + k];
            }
            ABf bV[4];
            #pragma unroll
            for (int nt = 0; nt < 4; ++nt) {
                const int h = nt * 16 + l15;
                const int kk = kb8 + ks * 32;
                #pragma unroll
                for (int p = 0; p < 8; ++p)
                    bV[nt].u[p] = *(const uint32_t*)&vT[h * 64 + kk + ((p >> 2) << 4) + ((p & 3) << 1)];
            }
            #pragma unroll
            for (int nt = 0; nt < 4; ++nt) accO[nt] = wmma_bf16(aP, bV[nt], accO[nt]);
        }
        __syncthreads();
    }

    const int rl = (lane >> 4) << 3, cl = lane & 15;
    #pragma unroll
    for (int nt = 0; nt < 4; ++nt)
        #pragma unroll
        for (int r = 0; r < 8; ++r) {
            const float l = rowL[wid * 16 + rl + r];
            const float o = accO[nt][r] / l;
            const int rg = tokbase + qblk * 128 + wid * 16 + rl + r;
            const int cg = head * 64 + nt * 16 + cl;
            AO[(size_t)rg * 768 + cg] = f2bf(o);
        }
}

__global__ __launch_bounds__(256) void scatter_kernel(
        const float* __restrict__ PIX, const int* __restrict__ meta,
        float* __restrict__ out) {
    int g = blockIdx.x;
    int idx = meta[4 * g], pw = meta[4 * g + 2], jj = meta[4 * g + 3];
    int py = jj / pw, px = jj % pw;
    #pragma unroll
    for (int i = 0; i < 3; ++i) {
        int k = threadIdx.x + i * 256;
        int c = k >> 8, dy = (k >> 4) & 15, dx = k & 15;
        out[((size_t)(idx * 3 + c) * 256 + py * 16 + dy) * 256 + px * 16 + dx] =
            PIX[(size_t)g * 768 + k];
    }
}

// ---------------------------------------------------------------- launch
extern "C" void kernel_launch(void* const* d_in, const int* in_sizes, int n_in,
                              void* d_out, int out_size, void* d_ws, size_t ws_size,
                              hipStream_t stream) {
    const float* images   = (const float*)d_in[0];
    const float* t        = (const float*)d_in[1];
    const int*   patch_hw = (const int*)d_in[2];
    const int*   packs    = (const int*)d_in[3];
    const float* conv_w   = (const float*)d_in[4];
    const float* conv_b   = (const float*)d_in[5];
    const float* ln1_w    = (const float*)d_in[6];
    const float* ln1_b    = (const float*)d_in[7];
    const float* in_w     = (const float*)d_in[8];
    const float* in_b     = (const float*)d_in[9];
    const float* out_w    = (const float*)d_in[10];
    const float* out_b    = (const float*)d_in[11];
    const float* ln2_w    = (const float*)d_in[12];
    const float* ln2_b    = (const float*)d_in[13];
    const float* mlp_w1   = (const float*)d_in[14];
    const float* mlp_b1   = (const float*)d_in[15];
    const float* mlp_w2   = (const float*)d_in[16];
    const float* mlp_b2   = (const float*)d_in[17];
    const float* lnf_w    = (const float*)d_in[18];
    const float* lnf_b    = (const float*)d_in[19];
    const float* pix_w    = (const float*)d_in[20];
    const float* pix_b    = (const float*)d_in[21];
    float* out = (float*)d_out;

    char* ws = (char*)d_ws;
    size_t off = 0;
    float*          X    = (float*)(ws + off);          off += (size_t)TOK * 768 * 4;
    unsigned short* XLN  = (unsigned short*)(ws + off); off += (size_t)TOK * 768 * 2;
    unsigned short* QKV  = (unsigned short*)(ws + off); off += (size_t)TOK * 2304 * 2;
    unsigned short* AO   = (unsigned short*)(ws + off); off += (size_t)TOK * 768 * 2;
    unsigned short* H    = (unsigned short*)(ws + off); off += (size_t)TOK * 3072 * 2;
    unsigned short* PATCH= (unsigned short*)(ws + off); off += (size_t)TOK * 768 * 2;
    float*          PIX  = (float*)(ws + off);          off += (size_t)TOK * 768 * 4;
    int*            META = (int*)(ws + off);            off += (size_t)TOK * 4 * 4;

    tok_meta_kernel<<<12, 256, 0, stream>>>(patch_hw, packs, META);
    zero_kernel<<<2048, 256, 0, stream>>>(out, 16 * 3 * 256 * 256);
    patch_gather_kernel<<<TOK, 256, 0, stream>>>(images, META, PATCH);
    gemm_bf16<0><<<dim3(6, 24), 256, 0, stream>>>(PATCH, conv_w, conv_b, X, nullptr,
                                                  TOK, 768, 768);
    embed_add_kernel<<<TOK, 256, 0, stream>>>(X, META, t);

    for (int l = 0; l < 8; ++l) {
        ln_kernel<<<TOK, 256, 0, stream>>>(X, ln1_w + l * 768, ln1_b + l * 768, XLN);
        gemm_bf16<1><<<dim3(18, 24), 256, 0, stream>>>(XLN, in_w + (size_t)l * 768 * 2304,
                                                       in_b + l * 2304, nullptr, QKV,
                                                       TOK, 2304, 768);
        attn_kernel<<<dim3(6, NHEAD, 4), 256, 0, stream>>>(QKV, AO);
        gemm_bf16<3><<<dim3(6, 24), 256, 0, stream>>>(AO, out_w + (size_t)l * 768 * 768,
                                                      out_b + l * 768, X, nullptr,
                                                      TOK, 768, 768);
        ln_kernel<<<TOK, 256, 0, stream>>>(X, ln2_w + l * 768, ln2_b + l * 768, XLN);
        gemm_bf16<2><<<dim3(24, 24), 256, 0, stream>>>(XLN, mlp_w1 + (size_t)l * 768 * 3072,
                                                       mlp_b1 + l * 3072, nullptr, H,
                                                       TOK, 3072, 768);
        gemm_bf16<3><<<dim3(6, 24), 256, 0, stream>>>(H, mlp_w2 + (size_t)l * 3072 * 768,
                                                      mlp_b2 + l * 768, X, nullptr,
                                                      TOK, 768, 3072);
    }

    ln_kernel<<<TOK, 256, 0, stream>>>(X, lnf_w, lnf_b, XLN);
    gemm_bf16<0><<<dim3(6, 24), 256, 0, stream>>>(XLN, pix_w, pix_b, PIX, nullptr,
                                                  TOK, 768, 768);
    scatter_kernel<<<TOK, 256, 0, stream>>>(PIX, META, out);
}
